// FactoredMoEProjection_77051713290693
// MI455X (gfx1250) — matrix operand order repaired and checked by reading, not compiled, not run on hardware
//
#include <hip/hip_runtime.h>
#include <hip/hip_bf16.h>

// ---------------- types ----------------
typedef __attribute__((ext_vector_type(16))) _Float16 v16h;
typedef __attribute__((ext_vector_type(8)))  _Float16 v8h;
typedef __attribute__((ext_vector_type(4)))  _Float16 v4h;
typedef __attribute__((ext_vector_type(8)))  float    v8f;

// Problem constants (from reference)
#define BATCH 32
#define CIN   512
#define HW    1024
#define NEXP  8
#define TOPK  2
#define RDIM  64
#define ODIM  512
#define BN_EPS 1e-5f

// LDS pitches (halves); all multiples of 8 halves (16B) for b128 alignment
#define HTP 136    // hT : 128 x 136
#define AP   72    // As : 128 x 72
#define XP   72    // xT : 128 x 72
#define UPITCH 136 // uA : 128 x 136

// LDS layout (bytes)
#define HT_BYTES   34816                    // 128*136*2
#define AS_BYTES   18432                    // 128*72*2
#define DOWNBUF    36864                    // As + xT
#define UA_BYTES   34816                    // 128*136*2
#define REG_BYTES  73728                    // 2 down bufs (up bufs alias inside)
#define SMEM_TOTAL (HT_BYTES + REG_BYTES)   // 108544

// ---------------- WMMA fragment loads (wave32 layouts per CDNA5 ISA 7.12.2) ----
// A 16x32 f16: lanes 0-15 row=lane, K = 0..7 (elems 0..7) and 16..23 (elems 8..15)
//              lanes 16-31 row=lane-16, K = 8..15 and 24..31
__device__ __forceinline__ v16h load_a_frag(const _Float16* base, int pitch, int ln, int hi) {
    const _Float16* p = base + ln * pitch + hi * 8;
    v8h s0 = *(const v8h*)(p);
    v8h s1 = *(const v8h*)(p + 16);
    return __builtin_shufflevector(s0, s1, 0,1,2,3,4,5,6,7,8,9,10,11,12,13,14,15);
}
// B 32x16 f16 (source stored n-major, k-contiguous): lanes 0-15 n=lane K=0..7,16..23,
// lanes 16-31 n=lane-16 K=8..15,24..31
__device__ __forceinline__ v16h load_b_frag(const _Float16* base, int pitch, int ln, int hi) {
    const _Float16* p = base + ln * pitch + hi * 8;
    v8h s0 = *(const v8h*)(p);
    v8h s1 = *(const v8h*)(p + 16);
    return __builtin_shufflevector(s0, s1, 0,1,2,3,4,5,6,7,8,9,10,11,12,13,14,15);
}

// ---------------- staging helpers (vectorized) ----------------
// As: 128 rows (stacked experts) x 64 c, f16 row-major.  float4 loads -> b64 stores.
// xT: x chunk transposed to [p][c] (c contiguous).  b32 loads -> b128 stores.
__device__ __forceinline__ void stage_down(const float* __restrict__ Wd,
                                           const float* __restrict__ xb,
                                           int e0, int e1, int kc,
                                           _Float16* __restrict__ As_,
                                           _Float16* __restrict__ xT_, int t) {
    #pragma unroll
    for (int i = 0; i < 8; ++i) {
        int idx = t + i * 256;               // 2048 float4 = 128 rows x 16
        int row = idx >> 4, c4 = idx & 15;
        int e = (row < 64) ? e0 : e1;
        int r = row & 63;
        const float4 v = *(const float4*)&Wd[((size_t)(e * RDIM + r)) * CIN + kc * 64 + c4 * 4];
        v4h hv;
        hv[0] = (_Float16)v.x; hv[1] = (_Float16)v.y;
        hv[2] = (_Float16)v.z; hv[3] = (_Float16)v.w;
        *(v4h*)&As_[row * AP + c4 * 4] = hv;
    }
    int p = t & 127, half = t >> 7;          // each thread: one pixel, 32 c's
    #pragma unroll
    for (int j = 0; j < 4; ++j) {
        int c0 = half * 32 + j * 8;
        v8h pk;
        #pragma unroll
        for (int u = 0; u < 8; ++u)
            pk[u] = (_Float16)xb[(size_t)(kc * 64 + c0 + u) * HW + p];
        *(v8h*)&xT_[p * XP + c0] = pk;
    }
}

// uA: 128 o-rows x 128 k (stacked experts), f16 with BN-scale*gate-weight folded in.
// Also computes per-o shift constant into cst (by threads < 128).
__device__ __forceinline__ void stage_up(const float* __restrict__ Wu,
                                         const float* __restrict__ gamma,
                                         const float* __restrict__ beta,
                                         const float* __restrict__ rmean,
                                         const float* __restrict__ rvar,
                                         int e0, int e1, float w0, float w1, int oc,
                                         _Float16* __restrict__ uA_,
                                         float* __restrict__ cst_, int t) {
    if (t < 128) {
        int oG = oc * 128 + t;
        float s0 = gamma[e0 * ODIM + oG] * rsqrtf(rvar[e0 * ODIM + oG] + BN_EPS);
        float s1 = gamma[e1 * ODIM + oG] * rsqrtf(rvar[e1 * ODIM + oG] + BN_EPS);
        cst_[t] = w0 * (beta[e0 * ODIM + oG] - rmean[e0 * ODIM + oG] * s0)
                + w1 * (beta[e1 * ODIM + oG] - rmean[e1 * ODIM + oG] * s1);
    }
    // per-thread fixed (k4, e, r); o varies with i -> scale recomputed (cache hits)
    int k4 = t & 31, o0 = t >> 5;
    int k = k4 * 4;
    int e, r; float wk;
    if (k < 64) { e = e0; r = k;      wk = w0; }
    else        { e = e1; r = k - 64; wk = w1; }
    #pragma unroll
    for (int i = 0; i < 16; ++i) {
        int o = o0 + i * 8;
        int oG = oc * 128 + o;
        float s = wk * gamma[e * ODIM + oG] * rsqrtf(rvar[e * ODIM + oG] + BN_EPS);
        const float4 v = *(const float4*)&Wu[((size_t)(e * ODIM + oG)) * RDIM + r];
        v4h hv;
        hv[0] = (_Float16)(s * v.x); hv[1] = (_Float16)(s * v.y);
        hv[2] = (_Float16)(s * v.z); hv[3] = (_Float16)(s * v.w);
        *(v4h*)&uA_[o * UPITCH + k] = hv;
    }
}

// ---------------- kernel 1: adaptive avg pool (B,C,H,W) -> (B,C) ----------------
__global__ __launch_bounds__(256)
void pool_kernel(const float* __restrict__ x, float* __restrict__ pool) {
    int bc = blockIdx.x;
    const float4* xp = (const float4*)(x + (size_t)bc * HW);
    float4 v = xp[threadIdx.x];              // 256 threads * 4 = 1024
    float s = v.x + v.y + v.z + v.w;
    #pragma unroll
    for (int off = 16; off > 0; off >>= 1) s += __shfl_down(s, off);
    __shared__ float wsum[8];
    if ((threadIdx.x & 31) == 0) wsum[threadIdx.x >> 5] = s;
    __syncthreads();
    if (threadIdx.x == 0) {
        float t = 0.f;
        #pragma unroll
        for (int i = 0; i < 8; ++i) t += wsum[i];
        pool[bc] = t * (1.0f / (float)HW);
    }
}

// ---------------- kernel 2: gate + aux losses ----------------
__global__ __launch_bounds__(256)
void gate_kernel(const float* __restrict__ pool, const float* __restrict__ Wg,
                 const float* __restrict__ bg,
                 int* __restrict__ gate_e, float* __restrict__ gate_w,
                 float* __restrict__ out_losses) {
    __shared__ float lg[BATCH][NEXP];
    __shared__ float pr[BATCH][NEXP];
    __shared__ float lse[BATCH];
    __shared__ int   se[BATCH][2];
    int t = threadIdx.x;
    {
        int b = t >> 3, e = t & 7;            // 256 = 32*8 pairs
        const float4* pb = (const float4*)(pool + b * CIN);
        const float4* we = (const float4*)(Wg + e * CIN);
        float s = 0.f;
        for (int i = 0; i < CIN / 4; ++i) {
            float4 a = pb[i], w = we[i];
            s += a.x * w.x + a.y * w.y + a.z * w.z + a.w * w.w;
        }
        lg[b][e] = s + bg[e];
    }
    __syncthreads();
    if (t < BATCH) {
        int b = t;
        float v0 = -1e30f, v1 = -1e30f; int i0 = 0, i1 = 0;
        for (int e = 0; e < NEXP; ++e) {
            float l = lg[b][e];
            if (l > v0) { v1 = v0; i1 = i0; v0 = l; i0 = e; }
            else if (l > v1) { v1 = l; i1 = e; }
        }
        float w1 = 1.0f / (1.0f + __expf(v0 - v1));   // softmax([v0,v1])[1]
        float w0 = 1.0f - w1;
        gate_e[2 * b] = i0; gate_e[2 * b + 1] = i1;
        gate_w[2 * b] = w0; gate_w[2 * b + 1] = w1;
        se[b][0] = i0; se[b][1] = i1;
        float m = v0, sum = 0.f;
        for (int e = 0; e < NEXP; ++e) { float p = __expf(lg[b][e] - m); pr[b][e] = p; sum += p; }
        float inv = 1.0f / sum;
        for (int e = 0; e < NEXP; ++e) pr[b][e] *= inv;
        lse[b] = m + __logf(sum);
    }
    __syncthreads();
    if (t == 0) {
        float imp[NEXP] = {0}, ld[NEXP] = {0};
        for (int b = 0; b < BATCH; ++b) {
            for (int e = 0; e < NEXP; ++e) imp[e] += pr[b][e];
            ld[se[b][0]] += 1.0f;
            ld[se[b][1]] += 1.0f;
        }
        float lbl = 0.f;
        for (int e = 0; e < NEXP; ++e) lbl += (imp[e] / (float)BATCH) * (ld[e] / (float)(BATCH * TOPK));
        lbl *= (float)NEXP;
        float z = 0.f;
        for (int b = 0; b < BATCH; ++b) z += lse[b] * lse[b];
        z /= (float)BATCH;
        out_losses[0] = lbl;
        out_losses[1] = z;
    }
}

// ---------------- kernel 3: fused expert GEMMs + SiLU + BN + weighted mix ------
// grid = (HW/128, B); block = 256 threads (8 waves, tiled 4M x 2N)
__global__ __launch_bounds__(256)
void moe_main(const float* __restrict__ x,  const float* __restrict__ Wd,
              const float* __restrict__ Wu, const float* __restrict__ gamma,
              const float* __restrict__ beta, const float* __restrict__ rmean,
              const float* __restrict__ rvar, const int* __restrict__ gate_e,
              const float* __restrict__ gate_w, float* __restrict__ out) {
    extern __shared__ char smem[];
    _Float16* hT = (_Float16*)smem;                     // [0, 34816)
    char* reg = smem + HT_BYTES;                        // 73728-byte region
    // down-phase double buffers: buf b at reg + b*DOWNBUF : As then xT
    // up-phase double buffers (alias): uA b at reg + b*UA_BYTES; cst at +69632
    float* cstbuf = (float*)(reg + 2 * UA_BYTES);       // 2 x 128 f32 (1024 B)

    const int t = threadIdx.x;
    const int b = blockIdx.y;
    const int pbase = blockIdx.x * 128;
    const int e0 = gate_e[2 * b], e1 = gate_e[2 * b + 1];
    const float w0 = gate_w[2 * b], w1 = gate_w[2 * b + 1];
    const int lane = t & 31, wave = t >> 5;
    const int wm = wave >> 1, wn = wave & 1;
    const int ln = lane & 15, hi = lane >> 4;

    const float* xb = x + (size_t)b * CIN * HW + pbase;

    // ---------- down-proj: h(128x128) = [Wd[e0];Wd[e1]](128x512) @ X(512x128) ----
    v8f acc[2][4];
    {
        v8f z = {};
        #pragma unroll
        for (int i = 0; i < 2; ++i)
            #pragma unroll
            for (int j = 0; j < 4; ++j) acc[i][j] = z;
    }

    stage_down(Wd, xb, e0, e1, 0, (_Float16*)reg, (_Float16*)(reg + AS_BYTES), t);

    for (int kc = 0; kc < 8; ++kc) {                    // C chunks of 64
        __syncthreads();                                // chunk kc staged; buf kc^1 free
        if (kc < 7) {
            char* nb = reg + ((kc + 1) & 1) * DOWNBUF;
            stage_down(Wd, xb, e0, e1, kc + 1, (_Float16*)nb, (_Float16*)(nb + AS_BYTES), t);
        }
        const char* cb = reg + (kc & 1) * DOWNBUF;
        const _Float16* As_ = (const _Float16*)cb;
        const _Float16* xT_ = (const _Float16*)(cb + AS_BYTES);
        #pragma unroll
        for (int ks = 0; ks < 64; ks += 32) {
            v16h a0 = load_a_frag(&As_[(wm * 32) * AP + ks], AP, ln, hi);
            v16h a1 = load_a_frag(&As_[(wm * 32 + 16) * AP + ks], AP, ln, hi);
            #pragma unroll
            for (int nt = 0; nt < 4; ++nt) {
                v16h bf = load_b_frag(&xT_[(wn * 64 + nt * 16) * XP + ks], XP, ln, hi);
                acc[0][nt] = __builtin_amdgcn_wmma_f32_16x16x32_f16(
                    false, a0, false, bf, (short)0, acc[0][nt], false, false);
                acc[1][nt] = __builtin_amdgcn_wmma_f32_16x16x32_f16(
                    false, a1, false, bf, (short)0, acc[1][nt], false, false);
            }
        }
    }

    // SiLU, convert to f16, store hT[p][r] (r contiguous) for up-proj B frags
    #pragma unroll
    for (int mt = 0; mt < 2; ++mt)
        #pragma unroll
        for (int nt = 0; nt < 4; ++nt) {
            v8f a = acc[mt][nt];
            int p = wn * 64 + nt * 16 + ln;
            int m0 = wm * 32 + mt * 16 + hi * 8;
            v8h hv;
            #pragma unroll
            for (int v = 0; v < 8; ++v) {
                float f = a[v];
                hv[v] = (_Float16)(f / (1.0f + __expf(-f)));  // silu
            }
            *(v8h*)&hT[p * HTP + m0] = hv;
        }

    // ---------- up-proj: Y(512x128) = [w0*s0*Wu[e0] | w1*s1*Wu[e1]](512x128) @ h ----
    stage_up(Wu, gamma, beta, rmean, rvar, e0, e1, w0, w1, 0,
             (_Float16*)reg, cstbuf, t);

    for (int oc = 0; oc < 4; ++oc) {                    // O chunks of 128
        __syncthreads();                                // uA/cst oc staged; hT visible
        if (oc < 3) {
            int nb = (oc + 1) & 1;
            stage_up(Wu, gamma, beta, rmean, rvar, e0, e1, w0, w1, oc + 1,
                     (_Float16*)(reg + nb * UA_BYTES), cstbuf + nb * 128, t);
        }
        const _Float16* uA_ = (const _Float16*)(reg + (oc & 1) * UA_BYTES);
        const float* cst_ = cstbuf + (oc & 1) * 128;

        v8f a2[2][4];
        {
            v8f z = {};
            #pragma unroll
            for (int i = 0; i < 2; ++i)
                #pragma unroll
                for (int j = 0; j < 4; ++j) a2[i][j] = z;
        }
        #pragma unroll
        for (int ks = 0; ks < 128; ks += 32) {
            v16h a0 = load_a_frag(&uA_[(wm * 32) * UPITCH + ks], UPITCH, ln, hi);
            v16h a1 = load_a_frag(&uA_[(wm * 32 + 16) * UPITCH + ks], UPITCH, ln, hi);
            #pragma unroll
            for (int nt = 0; nt < 4; ++nt) {
                v16h bf = load_b_frag(&hT[(wn * 64 + nt * 16) * HTP + ks], HTP, ln, hi);
                a2[0][nt] = __builtin_amdgcn_wmma_f32_16x16x32_f16(
                    false, a0, false, bf, (short)0, a2[0][nt], false, false);
                a2[1][nt] = __builtin_amdgcn_wmma_f32_16x16x32_f16(
                    false, a1, false, bf, (short)0, a2[1][nt], false, false);
            }
        }
        // add per-o shift and store out[b, oc*128 + o, pbase + p]
        #pragma unroll
        for (int mt = 0; mt < 2; ++mt)
            #pragma unroll
            for (int nt = 0; nt < 4; ++nt) {
                int p = wn * 64 + nt * 16 + ln;
                int m0 = wm * 32 + mt * 16 + hi * 8;
                float* op = out + ((size_t)b * ODIM + oc * 128 + m0) * HW + pbase + p;
                v8f a = a2[mt][nt];
                #pragma unroll
                for (int v = 0; v < 8; ++v)
                    op[(size_t)v * HW] = a[v] + cst_[m0 + v];
            }
    }
}

// ---------------- host entry ----------------
extern "C" void kernel_launch(void* const* d_in, const int* in_sizes, int n_in,
                              void* d_out, int out_size, void* d_ws, size_t ws_size,
                              hipStream_t stream) {
    (void)in_sizes; (void)n_in; (void)out_size; (void)ws_size;
    const float* x     = (const float*)d_in[0];
    const float* Wg    = (const float*)d_in[1];
    const float* bg    = (const float*)d_in[2];
    const float* Wd    = (const float*)d_in[3];
    const float* Wu    = (const float*)d_in[4];
    const float* gamma = (const float*)d_in[5];
    const float* beta  = (const float*)d_in[6];
    const float* rmean = (const float*)d_in[7];
    const float* rvar  = (const float*)d_in[8];
    float* out = (float*)d_out;

    // workspace layout
    float* pool   = (float*)d_ws;                             // 32*512 f32 = 65536 B
    int*   gate_e = (int*)((char*)d_ws + 65536);              // 64 ints
    float* gate_w = (float*)((char*)d_ws + 65536 + 256);      // 64 f32

    pool_kernel<<<BATCH * CIN, 256, 0, stream>>>(x, pool);
    gate_kernel<<<1, 256, 0, stream>>>(pool, Wg, bg, gate_e, gate_w,
                                       out + (size_t)BATCH * ODIM * HW);
    dim3 grid(HW / 128, BATCH);
    moe_main<<<grid, 256, SMEM_TOTAL, stream>>>(x, Wd, Wu, gamma, beta, rmean,
                                                rvar, gate_e, gate_w, out);
}
